// Qwen3Attention_70686571757840
// MI455X (gfx1250) — matrix operand order repaired and checked by reading, not compiled, not run on hardware
//
#include <hip/hip_runtime.h>
#include <hip/hip_bf16.h>

typedef _Float16 h16;
typedef __attribute__((ext_vector_type(16))) _Float16 v16h;
typedef __attribute__((ext_vector_type(8)))  _Float16 v8h;
typedef __attribute__((ext_vector_type(4)))  _Float16 v4h;
typedef __attribute__((ext_vector_type(2)))  _Float16 v2h;
typedef __attribute__((ext_vector_type(8)))  float    v8f;
typedef __attribute__((ext_vector_type(4)))  unsigned int u32x4;
typedef __attribute__((ext_vector_type(8)))  int          i32x8;
typedef __attribute__((ext_vector_type(4)))  int          i32x4;

#define T_SEQ 2048
#define HID   2048
#define NH    32
#define NKV   8
#define HD    128
#define QS    (NH*HD)        // 4096
#define KVS   (NKV*HD)       // 1024
#define NQKV  (QS + 2*KVS)   // 6144
#define SCALING 0.08838834764831845f   // 128^-0.5
#define NEG_BIG -1e30f
#define KT_STRIDE 136        // halfs; 128 data + 4-DWORD pad (bank-conflict-free)

__device__ __forceinline__ v8f wmma16(v16h a, v16h b, v8f c) {
  return __builtin_amdgcn_wmma_f32_16x16x32_f16(false, a, false, b, (short)0, c,
                                                false, false);
}

__device__ __forceinline__ v8f zero8() {
  v8f z;
#pragma unroll
  for (int i = 0; i < 8; ++i) z[i] = 0.f;
  return z;
}

__device__ __forceinline__ v16h pack2(v8h lo, v8h hi) {
  v16h r;
#pragma unroll
  for (int i = 0; i < 8; ++i) { r[i] = lo[i]; r[8 + i] = hi[i]; }
  return r;
}

__device__ __forceinline__ v16h a_from_f16(const h16* rowp, int k0, bool hi) {
  int ka = k0 + (hi ? 16 : 0);
  return pack2(*(const v8h*)(rowp + ka), *(const v8h*)(rowp + ka + 8));
}

__device__ __forceinline__ v16h b_from_f16(const h16* rowp, int k0, bool hi) {
  int kb = k0 + (hi ? 16 : 0);
  return pack2(*(const v8h*)(rowp + kb), *(const v8h*)(rowp + kb + 8));
}

// ---------------------------------------------------------------------------
// TDM: 2D f16 tile (tile_d0 inner elems x tile_d1 rows) from global to LDS,
// with LDS padding of 4 DWORDs after every 64 DWORDs (=> row stride 136 halfs
// for tile_d0 = 128). Tracked by TENSORcnt on the issuing wave.
// D# per CDNA5 ISA. 6-arg builtin form (clang-23).
// ---------------------------------------------------------------------------
__device__ __forceinline__ void tdm_load_k_tile(unsigned lds_addr, const h16* gptr,
                                                unsigned stride_elems) {
  unsigned long long ga = (unsigned long long)(uintptr_t)gptr;
  u32x4 g0;
  g0[0] = 1u;                                   // count=1, user mode, load
  g0[1] = lds_addr;                             // LDS byte address
  g0[2] = (unsigned)ga;                         // global_addr[31:0]
  g0[3] = (unsigned)((ga >> 32) & 0x01FFFFFFu)  // global_addr[56:32]
        | (2u << 30);                           // type = 2 ("image")
  i32x8 g1;
  g1[0] = (int)((1u << 16)      // data_size = 1 -> 2 bytes
              | (1u << 20)      // pad_enable
              | (5u << 22)      // pad_interval code 5 -> every 64 DWORDs
              | (3u << 25));    // pad_amount  code 3 -> 4 DWORDs (16B)
  unsigned td0 = 1024, td1 = 2048;              // tensor dims (OOB bounds)
  g1[1] = (int)(td0 << 16);                     // tensor_dim0[15:0] @ bits 63:48
  g1[2] = (int)((td0 >> 16) | (td1 << 16));     // dim0 hi | tensor_dim1 lo
  g1[3] = (int)((td1 >> 16) | (128u << 16));    // dim1 hi | tile_dim0 = 128
  g1[4] = (int)32u;                             // tile_dim1 = 32, tile_dim2 = 0
  g1[5] = (int)stride_elems;                    // tensor_dim0_stride[31:0]
  g1[6] = 0;                                    // stride hi | dim1_stride lo
  g1[7] = 0;
  i32x4 z4; z4[0] = 0; z4[1] = 0; z4[2] = 0; z4[3] = 0;
  i32x8 z8;
#pragma unroll
  for (int i = 0; i < 8; ++i) z8[i] = 0;
  __builtin_amdgcn_tensor_load_to_lds(g0, g1, z4, z4, z8, 0);
}

// ---------------------------------------------------------------------------
// Kernel 0: bulk fp32 -> f16 conversion.
// ---------------------------------------------------------------------------
__global__ __launch_bounds__(256)
void cvt_f16_kernel(const float* __restrict__ src, h16* __restrict__ dst, int n8)
{
  int i = blockIdx.x * 256 + threadIdx.x;
  if (i >= n8) return;
  const float4* p = (const float4*)(src + (size_t)i * 8);
  float4 x = p[0], y = p[1];
  v8h o;
  o[0] = (h16)x.x; o[1] = (h16)x.y; o[2] = (h16)x.z; o[3] = (h16)x.w;
  o[4] = (h16)y.x; o[5] = (h16)y.y; o[6] = (h16)y.z; o[7] = (h16)y.w;
  *(v8h*)(dst + (size_t)i * 8) = o;
}

// ---------------------------------------------------------------------------
// Kernel 1: qkv = hidden @ wqkv^T + fused per-head RMSNorm (q,k) + RoPE (q,k).
// ---------------------------------------------------------------------------
__global__ __launch_bounds__(128)
void qkv_rope_kernel(const h16* __restrict__ hsh, const h16* __restrict__ wqkvh,
                     const int* __restrict__ pos,
                     const float* __restrict__ qw, const float* __restrict__ kw,
                     h16* __restrict__ qo, h16* __restrict__ ko, h16* __restrict__ vo)
{
  const int t0   = blockIdx.x * 32;
  const int j0   = blockIdx.y * 128;
  const int tid  = threadIdx.x;
  const int wave = tid >> 5, lane = tid & 31, lh = lane & 15;
  const bool hi  = lane >= 16;
  const int n0   = wave * 32;

  __shared__ float tile[32][132];
  __shared__ float psum[32][4];
  __shared__ float scl[32];

  v8f acc00 = zero8(), acc01 = zero8(), acc10 = zero8(), acc11 = zero8();
  const h16* arow0 = hsh   + (size_t)(t0 + lh) * HID;
  const h16* arow1 = hsh   + (size_t)(t0 + 16 + lh) * HID;
  const h16* brow0 = wqkvh + (size_t)(j0 + n0 + lh) * HID;
  const h16* brow1 = wqkvh + (size_t)(j0 + n0 + 16 + lh) * HID;

  for (int k0 = 0; k0 < HID; k0 += 32) {
    v16h a0 = a_from_f16(arow0, k0, hi);
    v16h a1 = a_from_f16(arow1, k0, hi);
    v16h b0 = b_from_f16(brow0, k0, hi);
    v16h b1 = b_from_f16(brow1, k0, hi);
    acc00 = wmma16(a0, b0, acc00);
    acc01 = wmma16(a0, b1, acc01);
    acc10 = wmma16(a1, b0, acc10);
    acc11 = wmma16(a1, b1, acc11);
  }

#pragma unroll
  for (int i = 0; i < 8; ++i) {
    int m = i + (hi ? 8 : 0);
    tile[m][n0 + lh]           = acc00[i];
    tile[m][n0 + 16 + lh]      = acc01[i];
    tile[m + 16][n0 + lh]      = acc10[i];
    tile[m + 16][n0 + 16 + lh] = acc11[i];
  }
  __syncthreads();

  {
    int r = tid & 31, c0 = (tid >> 5) * 32;
    float s = 0.f;
#pragma unroll
    for (int c = 0; c < 32; ++c) { float v = tile[r][c0 + c]; s += v * v; }
    psum[r][tid >> 5] = s;
  }
  __syncthreads();
  if (tid < 32) {
    float s = psum[tid][0] + psum[tid][1] + psum[tid][2] + psum[tid][3];
    scl[tid] = rsqrtf(s * (1.f / 128.f) + 1e-6f);
  }
  __syncthreads();

  const int r = tid & 31, c0 = (tid >> 5) * 32;
  const int t = t0 + r;
  if (j0 < QS + KVS) {
    const float* w = (j0 < QS) ? qw : kw;
    h16* dst = (j0 < QS) ? (qo + (size_t)t * QS + j0)
                         : (ko + (size_t)t * KVS + (j0 - QS));
    const float sc = scl[r];
    const float p  = (float)pos[t];
#pragma unroll
    for (int c = 0; c < 32; ++c) {
      int d = c0 + c;
      int f = d & 63;
      float invf = __powf(10000.f, -(float)f * (1.f / 64.f));
      float sn, cs;
      __sincosf(p * invf, &sn, &cs);
      float x = tile[r][d] * sc * w[d];
      float y;
      if (d < 64) {
        float x2 = tile[r][d + 64] * sc * w[d + 64];
        y = x * cs - x2 * sn;
      } else {
        float x1 = tile[r][d - 64] * sc * w[d - 64];
        y = x * cs + x1 * sn;
      }
      dst[d] = (h16)y;
    }
  } else {
    h16* dst = vo + (size_t)t * KVS + (j0 - QS - KVS);
#pragma unroll
    for (int c = 0; c < 32; ++c) { int d = c0 + c; dst[d] = (h16)tile[r][d]; }
  }
}

// ---------------------------------------------------------------------------
// Kernel 2: causal flash attention. K tile DMA'd by the Tensor Data Mover
// into a double-buffered padded LDS tile (overlapped with compute, TENSORcnt
// synchronized); V staged transposed by vector path; per-wave probs staging
// for the score-D -> A re-layout.
// ---------------------------------------------------------------------------
__global__ __launch_bounds__(128)
void attn_kernel(const h16* __restrict__ qg, const h16* __restrict__ kg,
                 const h16* __restrict__ vg, h16* __restrict__ og)
{
  const int h    = blockIdx.y;
  const int kvh  = h >> 2;
  const int tid  = threadIdx.x;
  const int wave = tid >> 5, lane = tid & 31, lh = lane & 15;
  const bool hih = lane >= 16;
  const int t0   = blockIdx.x * 64 + wave * 16;

  __shared__ __attribute__((aligned(16))) h16 kt[2][32 * KT_STRIDE];
  __shared__ __attribute__((aligned(16))) h16 vt[HD * 40];
  __shared__ __attribute__((aligned(16))) h16 pt[4][16 * 40];

  v16h qa[4];
  const h16* qrow = qg + (size_t)(t0 + lh) * QS + h * HD;
#pragma unroll
  for (int j = 0; j < 4; ++j) qa[j] = a_from_f16(qrow, j * 32, hih);

  v8f oacc[8];
#pragma unroll
  for (int nt = 0; nt < 8; ++nt) oacc[nt] = zero8();
  float mrow[8], lrow[8];
#pragma unroll
  for (int i = 0; i < 8; ++i) { mrow[i] = NEG_BIG; lrow[i] = 0.f; }

  const int nChunks = blockIdx.x * 2 + 2;
  const int kvbase  = kvh * HD;

  // prologue: DMA chunk 0's K tile
  if (wave == 0)
    tdm_load_k_tile((unsigned)(uintptr_t)&kt[0][0], kg + kvbase, KVS);

  for (int ch = 0; ch < nChunks; ++ch) {
    const int s0 = ch * 32;
    __syncthreads();                            // prior readers of vt / kt[(ch+1)&1] done
    // V tile: 2(s) x 4(d) micro-tiles, transposed store as packed b32
#pragma unroll
    for (int it = 0; it < 4; ++it) {
      int c = it * 128 + tid;                   // 0..511
      int sp = c >> 5, dq = c & 31;
      int vs = 2 * sp, vd = dq * 4;
      v4h va = *(const v4h*)(vg + (size_t)(s0 + vs) * KVS + kvbase + vd);
      v4h vb = *(const v4h*)(vg + (size_t)(s0 + vs + 1) * KVS + kvbase + vd);
#pragma unroll
      for (int j2 = 0; j2 < 4; ++j2) {
        v2h pr; pr[0] = va[j2]; pr[1] = vb[j2];
        *(v2h*)&vt[(vd + j2) * 40 + vs] = pr;
      }
    }
    if (wave == 0) {
      if (ch + 1 < nChunks) {
        // issue next chunk's DMA into the other buffer, then retire current
        tdm_load_k_tile((unsigned)(uintptr_t)&kt[(ch + 1) & 1][0],
                        kg + (size_t)(s0 + 32) * KVS + kvbase, KVS);
        __builtin_amdgcn_s_wait_tensorcnt((short)1);
      } else {
        __builtin_amdgcn_s_wait_tensorcnt((short)0);
      }
    }
    if (ch + 1 < nChunks)
      __builtin_prefetch(vg + (size_t)(s0 + 32 + lane) * KVS + kvbase, 0, 1);
    __syncthreads();                            // publish V stores + K DMA

    if (s0 <= t0 + 15) {                        // wave-uniform: EXEC all-ones
      const h16* kbuf = &kt[ch & 1][0];
      v8f sacc[2]; sacc[0] = zero8(); sacc[1] = zero8();
#pragma unroll
      for (int nt = 0; nt < 2; ++nt) {
        int key = nt * 16 + lh;
#pragma unroll
        for (int j = 0; j < 4; ++j) {
          int ds = j * 32 + (hih ? 16 : 0);
          v16h b = pack2(*(const v8h*)&kbuf[key * KT_STRIDE + ds],
                         *(const v8h*)&kbuf[key * KT_STRIDE + ds + 8]);
          sacc[nt] = wmma16(qa[j], b, sacc[nt]);
        }
      }
      float alpha[8];
#pragma unroll
      for (int i = 0; i < 8; ++i) {
        int m = t0 + i + (hih ? 8 : 0);
        bool ok0 = (s0 + lh) <= m;
        bool ok1 = (s0 + 16 + lh) <= m;
        float v0 = ok0 ? sacc[0][i] * SCALING : NEG_BIG;
        float v1 = ok1 ? sacc[1][i] * SCALING : NEG_BIG;
        float cm = fmaxf(v0, v1);
        cm = fmaxf(cm, __shfl_xor(cm, 1, 32));
        cm = fmaxf(cm, __shfl_xor(cm, 2, 32));
        cm = fmaxf(cm, __shfl_xor(cm, 4, 32));
        cm = fmaxf(cm, __shfl_xor(cm, 8, 32));
        float mnew = fmaxf(mrow[i], cm);
        alpha[i] = __expf(mrow[i] - mnew);
        mrow[i]  = mnew;
        float p0 = ok0 ? __expf(v0 - mnew) : 0.f;
        float p1 = ok1 ? __expf(v1 - mnew) : 0.f;
        float rs = p0 + p1;
        rs += __shfl_xor(rs, 1, 32);
        rs += __shfl_xor(rs, 2, 32);
        rs += __shfl_xor(rs, 4, 32);
        rs += __shfl_xor(rs, 8, 32);
        lrow[i] = lrow[i] * alpha[i] + rs;
        sacc[0][i] = p0; sacc[1][i] = p1;
      }
      h16* pw = pt[wave];
#pragma unroll
      for (int i = 0; i < 8; ++i) {
        int m = i + (hih ? 8 : 0);
        pw[m * 40 + lh]      = (h16)sacc[0][i];
        pw[m * 40 + 16 + lh] = (h16)sacc[1][i];
      }
#pragma unroll
      for (int nt = 0; nt < 8; ++nt)
#pragma unroll
        for (int i = 0; i < 8; ++i) oacc[nt][i] *= alpha[i];
      {
        int ks = hih ? 16 : 0;
        v16h pa = pack2(*(const v8h*)&pw[lh * 40 + ks],
                        *(const v8h*)&pw[lh * 40 + ks + 8]);
        int sb = hih ? 16 : 0;
#pragma unroll
        for (int nt = 0; nt < 8; ++nt) {
          int d = nt * 16 + lh;
          v16h vb2 = pack2(*(const v8h*)&vt[d * 40 + sb],
                           *(const v8h*)&vt[d * 40 + sb + 8]);
          oacc[nt] = wmma16(pa, vb2, oacc[nt]);
        }
      }
    }
  }

#pragma unroll
  for (int nt = 0; nt < 8; ++nt) {
    int d = nt * 16 + lh;
#pragma unroll
    for (int i = 0; i < 8; ++i) {
      int m = i + (hih ? 8 : 0);
      og[(size_t)(t0 + m) * QS + h * HD + d] = (h16)(oacc[nt][i] / lrow[i]);
    }
  }
}

// ---------------------------------------------------------------------------
// Kernel 3: out = o @ wo^T (fp32 result).
// ---------------------------------------------------------------------------
__global__ __launch_bounds__(128)
void out_gemm_kernel(const h16* __restrict__ og, const h16* __restrict__ woh,
                     float* __restrict__ out)
{
  const int t0   = blockIdx.x * 32;
  const int tid  = threadIdx.x;
  const int wave = tid >> 5, lane = tid & 31, lh = lane & 15;
  const bool hi  = lane >= 16;
  const int n0   = blockIdx.y * 128 + wave * 32;

  v8f acc00 = zero8(), acc01 = zero8(), acc10 = zero8(), acc11 = zero8();
  const h16* arow0 = og  + (size_t)(t0 + lh) * QS;
  const h16* arow1 = og  + (size_t)(t0 + 16 + lh) * QS;
  const h16* brow0 = woh + (size_t)(n0 + lh) * QS;
  const h16* brow1 = woh + (size_t)(n0 + 16 + lh) * QS;

  for (int k0 = 0; k0 < QS; k0 += 32) {
    v16h a0 = a_from_f16(arow0, k0, hi);
    v16h a1 = a_from_f16(arow1, k0, hi);
    v16h b0 = b_from_f16(brow0, k0, hi);
    v16h b1 = b_from_f16(brow1, k0, hi);
    acc00 = wmma16(a0, b0, acc00);
    acc01 = wmma16(a0, b1, acc01);
    acc10 = wmma16(a1, b0, acc10);
    acc11 = wmma16(a1, b1, acc11);
  }
#pragma unroll
  for (int i = 0; i < 8; ++i) {
    int m = i + (hi ? 8 : 0);
    out[(size_t)(t0 + m) * HID + n0 + lh]           = acc00[i];
    out[(size_t)(t0 + m) * HID + n0 + 16 + lh]      = acc01[i];
    out[(size_t)(t0 + m + 16) * HID + n0 + lh]      = acc10[i];
    out[(size_t)(t0 + m + 16) * HID + n0 + 16 + lh] = acc11[i];
  }
}

// ---------------------------------------------------------------------------
extern "C" void kernel_launch(void* const* d_in, const int* in_sizes, int n_in,
                              void* d_out, int out_size, void* d_ws, size_t ws_size,
                              hipStream_t stream) {
  const int*   pos  = (const int*)d_in[0];
  const float* hs   = (const float*)d_in[1];
  const float* wqkv = (const float*)d_in[2];
  const float* wo   = (const float*)d_in[3];
  const float* qw   = (const float*)d_in[4];
  const float* kw   = (const float*)d_in[5];
  float* out = (float*)d_out;

  h16* hsh     = (h16*)d_ws;
  h16* wqkvh   = hsh   + (size_t)HID * T_SEQ;
  h16* woh     = wqkvh + (size_t)NQKV * HID;
  h16* wq      = woh   + (size_t)HID * QS;
  h16* wk      = wq    + (size_t)T_SEQ * QS;
  h16* wv      = wk    + (size_t)T_SEQ * KVS;
  h16* wo_attn = wv    + (size_t)T_SEQ * KVS;

  const int n8_hs   = (T_SEQ * HID) / 8;
  const int n8_wqkv = (NQKV * HID) / 8;
  const int n8_wo   = (HID * QS) / 8;
  cvt_f16_kernel<<<n8_hs   / 256, 256, 0, stream>>>(hs,   hsh,   n8_hs);
  cvt_f16_kernel<<<n8_wqkv / 256, 256, 0, stream>>>(wqkv, wqkvh, n8_wqkv);
  cvt_f16_kernel<<<n8_wo   / 256, 256, 0, stream>>>(wo,   woh,   n8_wo);

  dim3 blk(128);
  qkv_rope_kernel<<<dim3(T_SEQ / 32, NQKV / 128), blk, 0, stream>>>(
      hsh, wqkvh, pos, qw, kw, wq, wk, wv);
  attn_kernel<<<dim3(T_SEQ / 64, NH), blk, 0, stream>>>(wq, wk, wv, wo_attn);
  out_gemm_kernel<<<dim3(T_SEQ / 32, HID / 64), blk, 0, stream>>>(wo_attn, woh, out);
}
